// GraphTransformerBlock_9740985827991
// MI455X (gfx1250) — matrix operand order repaired and checked by reading, not compiled, not run on hardware
//
#include <hip/hip_runtime.h>
#include <hip/hip_bf16.h>
#include <math.h>

// ---------------- problem constants (match reference) ----------------
constexpr int NN = 50000;   // nodes
constexpr int DD = 128;     // model dim (= H*C)
constexpr int HH = 8;       // heads
constexpr int CC = 16;      // head dim
constexpr int EE = 800000;  // edges

typedef _Float16 v16h __attribute__((ext_vector_type(16)));
typedef _Float16 v8h  __attribute__((ext_vector_type(8)));
typedef _Float16 v4h  __attribute__((ext_vector_type(4)));
typedef float    v8f  __attribute__((ext_vector_type(8)));

// ---------------- helpers ----------------
__device__ __forceinline__ float wave_sum(float v) {
    v += __shfl_xor(v, 16, 32);
    v += __shfl_xor(v, 8, 32);
    v += __shfl_xor(v, 4, 32);
    v += __shfl_xor(v, 2, 32);
    v += __shfl_xor(v, 1, 32);
    return v;
}

__device__ __forceinline__ void atomic_add_f32(float* p, float v) {
    __hip_atomic_fetch_add(p, v, __ATOMIC_RELAXED, __HIP_MEMORY_SCOPE_AGENT);
}

// order-preserving float<->uint map for atomicMax-based segment max
__device__ __forceinline__ unsigned flip_f32(float f) {
    unsigned b = __float_as_uint(f);
    return (b & 0x80000000u) ? ~b : (b | 0x80000000u);
}
__device__ __forceinline__ float unflip_f32(unsigned u) {
    unsigned b = (u & 0x80000000u) ? (u & 0x7fffffffu) : ~u;
    return __uint_as_float(b);
}

// B-matrix pack index for v_wmma_f32_16x16x32_f16:
// lane l holds col n=l%16; K chunk of 32 split as [0..15] (lanes 0-15) / [16..31] (lanes 16-31)
__device__ __forceinline__ size_t bpack_idx(int k, int n, int ncols) {
    return ((size_t)(k >> 5) * ncols + n) * 32 + (size_t)(((k >> 4) & 1) * 16) + (k & 15);
}

// ---------------- weight conversion / packing ----------------
__global__ __launch_bounds__(256) void pack_weights_kernel(
    const float* __restrict__ wq, const float* __restrict__ wk,
    const float* __restrict__ wv, const float* __restrict__ wskip,
    const float* __restrict__ w1, const float* __restrict__ w2,
    const float* __restrict__ bq, const float* __restrict__ bk,
    const float* __restrict__ bv, const float* __restrict__ bskip,
    _Float16* __restrict__ pqkvs, _Float16* __restrict__ p1,
    _Float16* __restrict__ p2, float* __restrict__ biascat)
{
    int idx = blockIdx.x * 256 + threadIdx.x;
    if (idx < 65536) {                 // qkvs: K=128, N=512 (wq|wk|wv|wskip)
        int k = idx >> 9, n = idx & 511;
        int sel = n >> 7, nn = n & 127;
        float v;
        if      (sel == 0) v = wq  [k * 128 + nn];
        else if (sel == 1) v = wk  [k * 128 + nn];
        else if (sel == 2) v = wv  [k * 128 + nn];
        else               v = wskip[k * 128 + nn];
        pqkvs[bpack_idx(k, n, 512)] = (_Float16)v;
    } else if (idx < 98304) {          // w1: K=128, N=256
        int i = idx - 65536;
        int k = i >> 8, n = i & 255;
        p1[bpack_idx(k, n, 256)] = (_Float16)w1[k * 256 + n];
    } else if (idx < 131072) {         // w2: K=256, N=128
        int i = idx - 98304;
        int k = i >> 7, n = i & 127;
        p2[bpack_idx(k, n, 128)] = (_Float16)w2[k * 128 + n];
    } else if (idx < 131584) {         // bias concat [512]
        int i = idx - 131072;
        int sel = i >> 7, nn = i & 127;
        float v;
        if      (sel == 0) v = bq[nn];
        else if (sel == 1) v = bk[nn];
        else if (sel == 2) v = bv[nn];
        else               v = bskip[nn];
        biascat[i] = v;
    }
}

// ---------------- scratch init (out=0, m=flip(-inf), denom=0) ----------------
__global__ __launch_bounds__(256) void init_kernel(
    float* __restrict__ outattn, unsigned* __restrict__ m, float* __restrict__ denom)
{
    int idx = blockIdx.x * 256 + threadIdx.x;       // NN*128 threads
    outattn[idx] = 0.0f;
    if (idx < NN * HH) { m[idx] = 0x007FFFFFu; denom[idx] = 0.0f; }
}

// ---------------- LayerNorm -> f16 (wave per node) ----------------
__global__ __launch_bounds__(256) void ln_f16_kernel(
    const float* __restrict__ x, const float* __restrict__ g,
    const float* __restrict__ b, _Float16* __restrict__ out)
{
    int n = blockIdx.x * 8 + threadIdx.y;
    int lane = threadIdx.x;
    int c = lane * 4;
    float4 v = *(const float4*)(x + (size_t)n * DD + c);
    float mu = wave_sum(v.x + v.y + v.z + v.w) * (1.0f / 128.0f);
    float d0 = v.x - mu, d1 = v.y - mu, d2 = v.z - mu, d3 = v.w - mu;
    float var = wave_sum(d0 * d0 + d1 * d1 + d2 * d2 + d3 * d3) * (1.0f / 128.0f);
    float rn = rsqrtf(var + 1e-5f);
    float4 gg = *(const float4*)(g + c);
    float4 bb = *(const float4*)(b + c);
    v4h o;
    o[0] = (_Float16)(d0 * rn * gg.x + bb.x);
    o[1] = (_Float16)(d1 * rn * gg.y + bb.y);
    o[2] = (_Float16)(d2 * rn * gg.z + bb.z);
    o[3] = (_Float16)(d3 * rn * gg.w + bb.w);
    *(v4h*)(out + (size_t)n * DD + c) = o;
}

// ---------------- WMMA GEMM: out[M,NCOLS] = A_f16[M,KDIM] @ Bpack + bias ----------------
// MODE 0: store fp32                       (qkvs fused projection)
// MODE 1: exact gelu, store f16            (FFN up)
// MODE 2: add residual, store fp32         (FFN down -> d_out)
template<int KDIM, int NCOLS, int MODE>
__global__ __launch_bounds__(256) void gemm_wmma_kernel(
    const _Float16* __restrict__ A, const _Float16* __restrict__ Bp,
    const float* __restrict__ bias, const float* __restrict__ resid,
    float* __restrict__ outF, _Float16* __restrict__ outH)
{
    const int lane = threadIdx.x;            // 0..31
    const int wave = threadIdx.y;            // 0..7
    constexpr int colTiles = NCOLS / 16;
    const int tile = blockIdx.x * 8 + wave;
    const int rowTile = tile / colTiles;
    const int colTile = tile % colTiles;
    const int rowBase = rowTile * 16;
    const int colBase = colTile * 16;
    const int r = rowBase + (lane & 15);

    const _Float16* ap0 = A + (size_t)r * KDIM + ((lane < 16) ? 0 : 16);
    const _Float16* bp0 = Bp + ((size_t)colBase + (lane & 15)) * 32 + ((lane >> 4) << 4);

    v8f acc = {};
    #pragma unroll
    for (int kc = 0; kc < KDIM / 32; ++kc) {
        union { v16h v; v8h h[2]; } a;
        a.h[0] = *(const v8h*)(ap0 + kc * 32);        // K base   .. base+7
        a.h[1] = *(const v8h*)(ap0 + kc * 32 + 8);    // K base+8 .. base+15
        v16h bmat = *(const v16h*)(bp0 + (size_t)kc * NCOLS * 32);
        acc = __builtin_amdgcn_wmma_f32_16x16x32_f16(
            false, a.v, false, bmat, (short)0, acc, false, false);
    }

    // epilogue: lanes 0-15 hold rows rowBase+0..7 (VGPR i -> row i),
    // lanes 16-31 hold rows rowBase+8..15. Hoist lane-half select out of the
    // loop so stores become base + i*NCOLS immediate offsets.
    const int col = colBase + (lane & 15);
    const float bb = bias[col];
    const size_t ebase = (size_t)(rowBase + ((lane >> 4) << 3)) * NCOLS + col;

    if (MODE == 0) {
        float* po = outF + ebase;
        #pragma unroll
        for (int i = 0; i < 8; ++i)
            po[(size_t)i * NCOLS] = acc[i] + bb;
    } else if (MODE == 1) {
        _Float16* ph = outH + ebase;
        #pragma unroll
        for (int i = 0; i < 8; ++i) {
            float val = acc[i] + bb;
            float gel = 0.5f * val * (1.0f + erff(val * 0.70710678118654752f));
            ph[(size_t)i * NCOLS] = (_Float16)gel;
        }
    } else {
        float* po = outF + ebase;
        const float* pr = resid + ebase;
        #pragma unroll
        for (int i = 0; i < 8; ++i)
            po[(size_t)i * NCOLS] = acc[i] + bb + pr[(size_t)i * NCOLS];
    }
}

// ---------------- edge e-recompute helper (lane handles cols 4l..4l+3) ----------------
__device__ __forceinline__ void edge_e_cols(
    const float* __restrict__ eattr, const float* __restrict__ wes,
    int e, int c0, float ec[4])
{
    float ea[16];
    const float4* eap = (const float4*)(eattr + (size_t)e * 16);
    #pragma unroll
    for (int i = 0; i < 4; ++i) {
        float4 t = eap[i];
        ea[4 * i] = t.x; ea[4 * i + 1] = t.y; ea[4 * i + 2] = t.z; ea[4 * i + 3] = t.w;
    }
    ec[0] = ec[1] = ec[2] = ec[3] = 0.0f;
    #pragma unroll
    for (int d = 0; d < 16; ++d) {
        float a = ea[d];
        const float* w = wes + d * DD + c0;
        ec[0] += a * w[0]; ec[1] += a * w[1]; ec[2] += a * w[2]; ec[3] += a * w[3];
    }
}

// ---------------- edge pass A: alpha = (q[dst] . (k[src]+e)) / 4, segment max ----------------
__global__ __launch_bounds__(256) void edge_alpha_kernel(
    const int* __restrict__ ei, const float* __restrict__ eattr,
    const float* __restrict__ we, const float* __restrict__ qkvs,
    float* __restrict__ alpha, unsigned* __restrict__ m)
{
    __shared__ float wes[16 * DD];
    int t = threadIdx.y * 32 + threadIdx.x;
    for (int i = t; i < 16 * DD; i += 256) wes[i] = we[i];
    __syncthreads();

    int e = blockIdx.x * 8 + threadIdx.y;
    int lane = threadIdx.x;
    int src = ei[e], dst = ei[EE + e];
    int c0 = lane * 4;

    float ec[4];
    edge_e_cols(eattr, wes, e, c0, ec);

    float4 k4 = *(const float4*)(qkvs + (size_t)src * 512 + 128 + c0);
    float4 q4 = *(const float4*)(qkvs + (size_t)dst * 512 + c0);
    float p = q4.x * (k4.x + ec[0]) + q4.y * (k4.y + ec[1]) +
              q4.z * (k4.z + ec[2]) + q4.w * (k4.w + ec[3]);
    p += __shfl_xor(p, 1, 32);
    p += __shfl_xor(p, 2, 32);
    if ((lane & 3) == 0) {
        int h = lane >> 2;
        float a = p * 0.25f;   // 1/sqrt(C), C=16
        alpha[(size_t)e * HH + h] = a;
        atomicMax(m + (size_t)dst * HH + h, flip_f32(a));
    }
}

// ---------------- edge pass B: ea = exp(alpha - m[dst]); segment sum ----------------
__global__ __launch_bounds__(256) void edge_softmax_kernel(
    const int* __restrict__ ei, float* __restrict__ alpha,
    const unsigned* __restrict__ m, float* __restrict__ denom)
{
    int idx = blockIdx.x * 256 + threadIdx.x;      // EE*HH threads
    int e = idx >> 3, h = idx & 7;
    int dst = ei[EE + e];
    float mf = unflip_f32(m[(size_t)dst * HH + h]);
    float ea = expf(alpha[idx] - mf);
    alpha[idx] = ea;
    atomic_add_f32(denom + (size_t)dst * HH + h, ea);
}

// ---------------- edge pass C: scatter (v[src]+e) * alpha/denom into out[dst] ----------------
__global__ __launch_bounds__(256) void edge_msg_kernel(
    const int* __restrict__ ei, const float* __restrict__ eattr,
    const float* __restrict__ we, const float* __restrict__ qkvs,
    const float* __restrict__ alpha, const float* __restrict__ denom,
    float* __restrict__ outattn)
{
    __shared__ float wes[16 * DD];
    int t = threadIdx.y * 32 + threadIdx.x;
    for (int i = t; i < 16 * DD; i += 256) wes[i] = we[i];
    __syncthreads();

    int e = blockIdx.x * 8 + threadIdx.y;
    int lane = threadIdx.x;
    int src = ei[e], dst = ei[EE + e];
    int c0 = lane * 4;
    int h = lane >> 2;

    float ec[4];
    edge_e_cols(eattr, wes, e, c0, ec);

    float ea  = alpha[(size_t)e * HH + h];
    float den = denom[(size_t)dst * HH + h];
    float w = ea / (den + 1e-16f);

    float4 v4 = *(const float4*)(qkvs + (size_t)src * 512 + 256 + c0);
    float* op = outattn + (size_t)dst * DD + c0;
    atomic_add_f32(op + 0, (v4.x + ec[0]) * w);
    atomic_add_f32(op + 1, (v4.y + ec[1]) * w);
    atomic_add_f32(op + 2, (v4.z + ec[2]) * w);
    atomic_add_f32(op + 3, (v4.w + ec[3]) * w);
}

// ---------------- beta gate + residual + LN2 -> f16 (wave per node) ----------------
__global__ __launch_bounds__(256) void node_finish_kernel(
    const float* __restrict__ x, const float* __restrict__ qkvs,
    const float* __restrict__ outattn, const float* __restrict__ wbeta,
    const float* __restrict__ g2, const float* __restrict__ b2,
    float* __restrict__ x1, _Float16* __restrict__ hh)
{
    int n = blockIdx.x * 8 + threadIdx.y;
    int lane = threadIdx.x;
    int c = lane * 4;

    float4 o4 = *(const float4*)(outattn + (size_t)n * DD + c);
    float4 r4 = *(const float4*)(qkvs + (size_t)n * 512 + 384 + c);
    float4 w0 = *(const float4*)(wbeta + c);
    float4 w1 = *(const float4*)(wbeta + 128 + c);
    float4 w2 = *(const float4*)(wbeta + 256 + c);

    float sp = w0.x * o4.x + w1.x * r4.x + w2.x * (o4.x - r4.x)
             + w0.y * o4.y + w1.y * r4.y + w2.y * (o4.y - r4.y)
             + w0.z * o4.z + w1.z * r4.z + w2.z * (o4.z - r4.z)
             + w0.w * o4.w + w1.w * r4.w + w2.w * (o4.w - r4.w);
    float s = wave_sum(sp);
    float beta = 1.0f / (1.0f + expf(-s));
    float ob = 1.0f - beta;

    float4 xv = *(const float4*)(x + (size_t)n * DD + c);
    float y0 = xv.x + beta * r4.x + ob * o4.x;
    float y1 = xv.y + beta * r4.y + ob * o4.y;
    float y2 = xv.z + beta * r4.z + ob * o4.z;
    float y3 = xv.w + beta * r4.w + ob * o4.w;
    float4 y4 = {y0, y1, y2, y3};
    *(float4*)(x1 + (size_t)n * DD + c) = y4;

    // LayerNorm 2
    float mu = wave_sum(y0 + y1 + y2 + y3) * (1.0f / 128.0f);
    float d0 = y0 - mu, d1 = y1 - mu, d2 = y2 - mu, d3 = y3 - mu;
    float var = wave_sum(d0 * d0 + d1 * d1 + d2 * d2 + d3 * d3) * (1.0f / 128.0f);
    float rn = rsqrtf(var + 1e-5f);
    float4 gg = *(const float4*)(g2 + c);
    float4 bb = *(const float4*)(b2 + c);
    v4h o;
    o[0] = (_Float16)(d0 * rn * gg.x + bb.x);
    o[1] = (_Float16)(d1 * rn * gg.y + bb.y);
    o[2] = (_Float16)(d2 * rn * gg.z + bb.z);
    o[3] = (_Float16)(d3 * rn * gg.w + bb.w);
    *(v4h*)(hh + (size_t)n * DD + c) = o;
}

// ---------------- host side ----------------
extern "C" void kernel_launch(void* const* d_in, const int* in_sizes, int n_in,
                              void* d_out, int out_size, void* d_ws, size_t ws_size,
                              hipStream_t stream)
{
    (void)in_sizes; (void)n_in; (void)out_size; (void)ws_size;
    const float* x      = (const float*)d_in[0];
    const int*   ei     = (const int*)  d_in[1];
    const float* eattr  = (const float*)d_in[2];
    const float* wq     = (const float*)d_in[3];
    const float* bq     = (const float*)d_in[4];
    const float* wk     = (const float*)d_in[5];
    const float* bk     = (const float*)d_in[6];
    const float* wv     = (const float*)d_in[7];
    const float* bv     = (const float*)d_in[8];
    const float* we     = (const float*)d_in[9];
    const float* wskip  = (const float*)d_in[10];
    const float* bskip  = (const float*)d_in[11];
    const float* wbeta  = (const float*)d_in[12];
    const float* ln1g   = (const float*)d_in[13];
    const float* ln1b   = (const float*)d_in[14];
    const float* ln2g   = (const float*)d_in[15];
    const float* ln2b   = (const float*)d_in[16];
    const float* w1     = (const float*)d_in[17];
    const float* b1     = (const float*)d_in[18];
    const float* w2     = (const float*)d_in[19];
    const float* b2     = (const float*)d_in[20];
    float* out = (float*)d_out;

    char* ws = (char*)d_ws;
    // workspace layout (all 256B-aligned)
    const size_t off_xn     = 0;                              // N*128 f16   = 12.8 MB
    const size_t off_qkvs   = off_xn     + (size_t)NN*128*2;  // N*512 f32   = 102.4 MB
    const size_t off_pqkvs  = off_qkvs   + (size_t)NN*512*4;  // 128*512 f16
    const size_t off_p1     = off_pqkvs  + (size_t)128*512*2; // 128*256 f16
    const size_t off_p2     = off_p1     + (size_t)128*256*2; // 256*128 f16
    const size_t off_bias   = off_p2     + (size_t)256*128*2; // 512 f32
    const size_t off_alpha  = off_bias   + 2048;              // E*8 f32     = 25.6 MB
    const size_t off_m      = off_alpha  + (size_t)EE*HH*4;   // N*8 u32
    const size_t off_denom  = off_m      + (size_t)NN*HH*4;   // N*8 f32
    const size_t off_outatt = off_denom  + (size_t)NN*HH*4;   // N*128 f32   = 25.6 MB
    const size_t off_x1     = off_outatt + (size_t)NN*128*4;  // N*128 f32
    const size_t off_hh     = off_x1     + (size_t)NN*128*4;  // N*128 f16
    // h1 (FFN hidden, N*256 f16 = 25.6 MB) aliases alpha: edge phase fully done by then
    const size_t off_h1     = off_alpha;

    _Float16* xn    = (_Float16*)(ws + off_xn);
    float*    qkvs  = (float*)   (ws + off_qkvs);
    _Float16* pqkvs = (_Float16*)(ws + off_pqkvs);
    _Float16* p1    = (_Float16*)(ws + off_p1);
    _Float16* p2    = (_Float16*)(ws + off_p2);
    float*    bcat  = (float*)   (ws + off_bias);
    float*    alpha = (float*)   (ws + off_alpha);
    unsigned* mbuf  = (unsigned*)(ws + off_m);
    float*    denom = (float*)   (ws + off_denom);
    float*    outat = (float*)   (ws + off_outatt);
    float*    x1    = (float*)   (ws + off_x1);
    _Float16* hh    = (_Float16*)(ws + off_hh);
    _Float16* h1    = (_Float16*)(ws + off_h1);

    dim3 wblk(32, 8);

    // 1) pack weights to f16 (WMMA B layout) + bias concat
    pack_weights_kernel<<<514, 256, 0, stream>>>(wq, wk, wv, wskip, w1, w2,
                                                 bq, bk, bv, bskip,
                                                 pqkvs, p1, p2, bcat);
    // 2) init scatter targets
    init_kernel<<<NN * 128 / 256, 256, 0, stream>>>(outat, mbuf, denom);
    // 3) LN1 -> f16
    ln_f16_kernel<<<NN / 8, wblk, 0, stream>>>(x, ln1g, ln1b, xn);
    // 4) fused q|k|v|skip projection via WMMA: [N,128] x [128,512]
    gemm_wmma_kernel<128, 512, 0><<<(NN / 16) * (512 / 16) / 8, wblk, 0, stream>>>(
        xn, pqkvs, bcat, nullptr, qkvs, nullptr);
    // 5) edge attention logits + segment max
    edge_alpha_kernel<<<EE / 8, wblk, 0, stream>>>(ei, eattr, we, qkvs, alpha, mbuf);
    // 6) exp + segment sum
    edge_softmax_kernel<<<EE * HH / 256, 256, 0, stream>>>(ei, alpha, mbuf, denom);
    // 7) weighted message scatter
    edge_msg_kernel<<<EE / 8, wblk, 0, stream>>>(ei, eattr, we, qkvs, alpha, denom, outat);
    // 8) beta gate + residual + LN2 -> f16
    node_finish_kernel<<<NN / 8, wblk, 0, stream>>>(x, qkvs, outat, wbeta, ln2g, ln2b, x1, hh);
    // 9) FFN up + exact gelu via WMMA: [N,128] x [128,256] -> f16
    gemm_wmma_kernel<128, 256, 1><<<(NN / 16) * (256 / 16) / 8, wblk, 0, stream>>>(
        hh, p1, b1, nullptr, nullptr, h1);
    // 10) FFN down + residual via WMMA: [N,256] x [256,128] -> d_out
    gemm_wmma_kernel<256, 128, 2><<<(NN / 16) * (128 / 16) / 8, wblk, 0, stream>>>(
        h1, p2, b2, x1, out, nullptr);
}
